// MaskedImageInpaintingTransformer_81896436400428
// MI455X (gfx1250) — matrix operand order repaired and checked by reading, compiled, not run on hardware
//
#include <hip/hip_runtime.h>

// ---------------------------------------------------------------------------
// Swin shifted-window attention block for MI455X (gfx1250), wave32 + WMMA bf16
//   B=4, H=W=256, C=256, heads=8, hd=32, window=8 (N=64), shift=4
// ---------------------------------------------------------------------------

typedef __attribute__((ext_vector_type(16))) __bf16 v16bf;
typedef __attribute__((ext_vector_type(8)))  float  v8f;

__device__ __forceinline__ unsigned short f2bf(float f) {
    unsigned u = __float_as_uint(f);
    u += 0x7fffu + ((u >> 16) & 1u);          // round-to-nearest-even
    return (unsigned short)(u >> 16);
}

union FragU { v16bf v; unsigned u[8]; };

// A-matrix 16x32 bf16 fragment (contiguous K order used here, matches frag_b): lane holds row M=lane&15,
// element pair p -> K = hi*16+2p, hi = lane>>4.
// `row` points at the start of this lane's row (bf16, row-major), already
// offset by the k-step base.
__device__ __forceinline__ v16bf frag_a(const unsigned short* row, int hi) {
    FragU f;
#pragma unroll
    for (int p = 0; p < 8; ++p) {
        int k0 = hi * 16 + 2 * p;
        f.u[p] = *(const unsigned*)(row + k0);
    }
    return f.v;
}

// B-matrix 32x16 bf16 fragment: lane holds col N=lane&15, elements are 16
// contiguous K values starting at (kbase + hi*16). Caller passes the fully
// resolved base pointer (16 contiguous bf16).
__device__ __forceinline__ v16bf frag_b(const unsigned short* base) {
    FragU f;
    const unsigned* p = (const unsigned*)base;
#pragma unroll
    for (int i = 0; i < 8; ++i) f.u[i] = p[i];
    return f.v;
}

__device__ __forceinline__ v8f wmma_bf16(v16bf a, v16bf b, v8f c) {
    return __builtin_amdgcn_wmma_f32_16x16x32_bf16(
        /*neg_a=*/false, a, /*neg_b=*/false, b,
        /*c_mod=*/(short)0, c, /*reuse_a=*/false, /*reuse_b=*/false);
}

__device__ __forceinline__ int region_id(int ys, int xs) {
    int rr = (ys < 248) ? 0 : (ys < 252) ? 1 : 2;
    int rc = (xs < 248) ? 0 : (xs < 252) ? 1 : 2;
    return rr * 3 + rc;
}

// ---------------------------------------------------------------------------
// Kernel 0: convert weights fp32 -> bf16 (qkv_w 768x256, proj_w 256x256)
// ---------------------------------------------------------------------------
__global__ void cvt_weights_kernel(const float* __restrict__ qkvw,
                                   const float* __restrict__ projw,
                                   unsigned short* __restrict__ wq,
                                   unsigned short* __restrict__ wp) {
    int g = blockIdx.x * 256 + threadIdx.x;
    if (g < 768 * 256) wq[g] = f2bf(qkvw[g]);
    if (g < 256 * 256) wp[g] = f2bf(projw[g]);
}

// ---------------------------------------------------------------------------
// Kernel 1: per-window fused  shift-gather -> QKV GEMM -> masked softmax
//           attention -> reverse-shift scatter of per-head outputs to d_out.
// One block = one window (64 pixels), 8 waves, wave w = head w.
// Dynamic LDS: [0,96K) per-wave q/k/vT (12KB each), [96K,160K) x-tile then
// per-wave attention scratch (8KB each).
// ---------------------------------------------------------------------------
__global__ __launch_bounds__(256) void swin_attn_qkv_kernel(
    const float* __restrict__ x, const unsigned short* __restrict__ wqkv,
    float* __restrict__ out) {
    extern __shared__ char smem[];
    const int tid  = threadIdx.x;
    const int wave = tid >> 5, lane = tid & 31;
    const int hi   = lane >> 4, ln = lane & 15;
    const int bid  = blockIdx.x;
    const int b = bid >> 10, win = bid & 1023, wy = win >> 5, wx = win & 31;

    unsigned short* q_s = (unsigned short*)smem + wave * 6144;  // 64x32
    unsigned short* k_s = q_s + 2048;                           // 64x32
    unsigned short* vT  = q_s + 4096;                           // 32x64 (d-major)
    unsigned short* attn_base = (unsigned short*)(smem + 96 * 1024);
    unsigned short* x_s    = attn_base;                 // 64x256 (phase 1/2 only)
    unsigned short* attn_s = attn_base + wave * 4096;   // 64x64 per wave

    // ---- phase 1: stage shift-gathered window (x is pre-shift; roll +4,+4) --
#pragma unroll
    for (int it = 0; it < 16; ++it) {
        int r = it * 4 + (tid >> 6);          // window element 0..63
        int c = (tid & 63) * 4;               // channel group
        int ys = wy * 8 + (r >> 3), xs = wx * 8 + (r & 7);
        int ysrc = (ys + 252) & 255, xsrc = (xs + 252) & 255;
        const float4 xv =
            *(const float4*)(x + (((size_t)b * 256 + ysrc) * 256 + xsrc) * 256 + c);
        *(unsigned*)(x_s + r * 256 + c) =
            (unsigned)f2bf(xv.x) | ((unsigned)f2bf(xv.y) << 16);
        *(unsigned*)(x_s + r * 256 + c + 2) =
            (unsigned)f2bf(xv.z) | ((unsigned)f2bf(xv.w) << 16);
    }
    __syncthreads();

    // ---- phase 2: QKV GEMM for this head: 64 x 96 = 64 x 256 * 256 x 96 ----
    const int head = wave;
    v8f zero = {};
#pragma unroll
    for (int nt = 0; nt < 6; ++nt) {  // q0 q1 k0 k1 v0 v1 column tiles
        int obase = (nt < 2) ? (head * 32 + nt * 16)
                  : (nt < 4) ? (256 + head * 32 + (nt - 2) * 16)
                             : (512 + head * 32 + (nt - 4) * 16);
        v8f acc[4] = {zero, zero, zero, zero};
#pragma unroll
        for (int kk = 0; kk < 8; ++kk) {
            v16bf bf = frag_b(wqkv + (size_t)(obase + ln) * 256 + kk * 32 + hi * 16);
#pragma unroll
            for (int i = 0; i < 4; ++i) {
                v16bf af = frag_a(x_s + (i * 16 + ln) * 256 + kk * 32, hi);
                acc[i] = wmma_bf16(af, bf, acc[i]);
            }
        }
#pragma unroll
        for (int i = 0; i < 4; ++i) {
#pragma unroll
            for (int jv = 0; jv < 8; ++jv) {
                int m = i * 16 + hi * 8 + jv;
                unsigned short hv = f2bf(acc[i][jv]);
                if (nt < 2)      q_s[m * 32 + nt * 16 + ln] = hv;
                else if (nt < 4) k_s[m * 32 + (nt - 2) * 16 + ln] = hv;
                else             vT[((nt - 4) * 16 + ln) * 64 + m] = hv;  // transposed
            }
        }
    }
    __syncthreads();  // x_s region is reused as attn scratch below

    // ---- phase 3: attention (wave-private) --------------------------------
    int regk[4];
#pragma unroll
    for (int j = 0; j < 4; ++j) {
        int m = j * 16 + ln;
        regk[j] = region_id(wy * 8 + (m >> 3), wx * 8 + (m & 7));
    }

    v8f a44[4][4];
    {
        v16bf qf[4];
#pragma unroll
        for (int i = 0; i < 4; ++i) qf[i] = frag_a(q_s + (i * 16 + ln) * 32, hi);
#pragma unroll
        for (int j = 0; j < 4; ++j) {
            v16bf kf = frag_b(k_s + (j * 16 + ln) * 32 + hi * 16);
#pragma unroll
            for (int i = 0; i < 4; ++i) a44[i][j] = wmma_bf16(qf[i], kf, zero);
        }
    }

    const float SCALE = 0.17677669529663687f;  // 32^-0.5
#pragma unroll
    for (int i = 0; i < 4; ++i) {
#pragma unroll
        for (int jv = 0; jv < 8; ++jv) {
            int n = i * 16 + hi * 8 + jv;  // query index (row lives in one half)
            int regq = region_id(wy * 8 + (n >> 3), wx * 8 + (n & 7));
            float v[4];
#pragma unroll
            for (int j = 0; j < 4; ++j)
                v[j] = a44[i][j][jv] * SCALE + ((regq == regk[j]) ? 0.f : -100.f);
            float mx = fmaxf(fmaxf(v[0], v[1]), fmaxf(v[2], v[3]));
#pragma unroll
            for (int d = 1; d < 16; d <<= 1) mx = fmaxf(mx, __shfl_xor(mx, d, 16));
            float e[4], s = 0.f;
#pragma unroll
            for (int j = 0; j < 4; ++j) { e[j] = __expf(v[j] - mx); s += e[j]; }
#pragma unroll
            for (int d = 1; d < 16; d <<= 1) s += __shfl_xor(s, d, 16);
            float inv = 1.0f / s;
#pragma unroll
            for (int j = 0; j < 4; ++j)
                attn_s[n * 64 + j * 16 + ln] = f2bf(e[j] * inv);
        }
    }

    // ---- P . V and reverse-shift scatter ----------------------------------
#pragma unroll
    for (int i = 0; i < 4; ++i) {
#pragma unroll
        for (int dt = 0; dt < 2; ++dt) {
            v8f oc = zero;
#pragma unroll
            for (int kk = 0; kk < 2; ++kk) {
                v16bf af = frag_a(attn_s + (i * 16 + ln) * 64 + kk * 32, hi);
                v16bf bf = frag_b(vT + (dt * 16 + ln) * 64 + kk * 32 + hi * 16);
                oc = wmma_bf16(af, bf, oc);
            }
#pragma unroll
            for (int jv = 0; jv < 8; ++jv) {
                int n = i * 16 + hi * 8 + jv;
                int ys = wy * 8 + (n >> 3), xs = wx * 8 + (n & 7);
                int yd = (ys + 252) & 255, xd = (xs + 252) & 255;  // roll(-4,-4)
                out[(((size_t)b * 256 + yd) * 256 + xd) * 256 + head * 32 +
                    dt * 16 + ln] = oc[jv];
            }
        }
    }
}

// ---------------------------------------------------------------------------
// Kernel 2: out-projection, IN PLACE on d_out. Each block owns 128 rows
// exclusively (read + write), so staging rows to LDS then overwriting is safe.
// out[m][o] = sum_c o[m][c] * proj_w[o][c] + pb[o]
// ---------------------------------------------------------------------------
__global__ __launch_bounds__(256) void swin_proj_kernel(
    float* __restrict__ io, const unsigned short* __restrict__ pw,
    const float* __restrict__ pb) {
    extern __shared__ char smem[];
    unsigned short* a_s = (unsigned short*)smem;  // 128 x 256 bf16
    const int tid  = threadIdx.x;
    const int wave = tid >> 5, lane = tid & 31;
    const int hi   = lane >> 4, ln = lane & 15;
    const size_t m0 = (size_t)blockIdx.x * 128;

    // stage this block's 128 rows (fp32 -> bf16)
#pragma unroll
    for (int it = 0; it < 32; ++it) {
        int f   = it * 256 + tid;    // float4 index, 8192 total
        int row = f >> 6;            // 64 float4 per row
        int cq  = (f & 63) * 4;
        const float4 v = *(const float4*)(io + (m0 + row) * 256 + cq);
        *(unsigned*)(a_s + row * 256 + cq) =
            (unsigned)f2bf(v.x) | ((unsigned)f2bf(v.y) << 16);
        *(unsigned*)(a_s + row * 256 + cq + 2) =
            (unsigned)f2bf(v.z) | ((unsigned)f2bf(v.w) << 16);
    }
    __syncthreads();

    v8f zero = {};
    v8f acc[16];
#pragma unroll
    for (int nt = 0; nt < 16; ++nt) acc[nt] = zero;
#pragma unroll
    for (int kk = 0; kk < 8; ++kk) {
        v16bf af = frag_a(a_s + (wave * 16 + ln) * 256 + kk * 32, hi);
#pragma unroll
        for (int nt = 0; nt < 16; ++nt) {
            v16bf bf = frag_b(pw + (size_t)(nt * 16 + ln) * 256 + kk * 32 + hi * 16);
            acc[nt] = wmma_bf16(af, bf, acc[nt]);
        }
    }
#pragma unroll
    for (int nt = 0; nt < 16; ++nt) {
        float bias = pb[nt * 16 + ln];
#pragma unroll
        for (int jv = 0; jv < 8; ++jv) {
            int m = wave * 16 + hi * 8 + jv;
            io[(m0 + m) * 256 + nt * 16 + ln] = acc[nt][jv] + bias;
        }
    }
}

// ---------------------------------------------------------------------------
extern "C" void kernel_launch(void* const* d_in, const int* in_sizes, int n_in,
                              void* d_out, int out_size, void* d_ws,
                              size_t ws_size, hipStream_t stream) {
    (void)in_sizes; (void)n_in; (void)out_size; (void)ws_size;
    const float* x     = (const float*)d_in[0];
    const float* qkvw  = (const float*)d_in[1];
    const float* projw = (const float*)d_in[2];
    const float* pb    = (const float*)d_in[3];

    unsigned short* wq = (unsigned short*)d_ws;   // 768*256 bf16
    unsigned short* wp = wq + 768 * 256;          // 256*256 bf16
    float* out = (float*)d_out;

    hipFuncSetAttribute((const void*)swin_attn_qkv_kernel,
                        hipFuncAttributeMaxDynamicSharedMemorySize, 163840);
    hipFuncSetAttribute((const void*)swin_proj_kernel,
                        hipFuncAttributeMaxDynamicSharedMemorySize, 65536);

    cvt_weights_kernel<<<768, 256, 0, stream>>>(qkvw, projw, wq, wp);
    swin_attn_qkv_kernel<<<4096, 256, 163840, stream>>>(x, wq, out);
    swin_proj_kernel<<<2048, 256, 65536, stream>>>(out, wp, pb);
}
